// NeuromodulatedBlock_35940286333494
// MI455X (gfx1250) — matrix operand order repaired and checked by reading, not compiled, not run on hardware
//
#include <hip/hip_runtime.h>
#include <hip/hip_bf16.h>

// ---------------------------------------------------------------------------
// NeuromodulatedBlock for MI455X (gfx1250, wave32, WMMA + async Global->LDS)
//   x_norm = x / max(||x||, eps)
//   hidden = x_norm @ W_enc^T           (bf16x3 WMMA GEMM, fp32 accum)
//   act    = topk_mask(hidden) * hidden (radix-select threshold per row)
//   out    = x + gate * (act @ W_dec^T) (bf16x3 WMMA GEMM, fused epilogue)
// GEMM stages K-slabs into LDS with global_load_async_to_lds_b128 (ASYNCcnt),
// double-buffered so DMA of slab t+1 overlaps WMMA on slab t.
// ---------------------------------------------------------------------------

#define B_DIM   4096
#define IN_DIM  1024
#define HID_DIM 8192
#define K_SEL   256

#define BM 64          // block tile M
#define BN 128         // block tile N
#define BK 32          // K step
#define LDS_ROW 40     // padded bf16 elems per 32-elem row (bank-conflict free)

typedef __attribute__((ext_vector_type(16))) __bf16 v16bf;
typedef __attribute__((ext_vector_type(8)))  __bf16 v8bf;
typedef __attribute__((ext_vector_type(8)))  float  v8f;

// ---------------------------------------------------------------------------
// Async Global -> LDS copy of one 16-byte chunk per lane (ASYNCcnt-tracked)
// ---------------------------------------------------------------------------
__device__ __forceinline__ void async_b128(unsigned lds_off, unsigned long long gaddr) {
    asm volatile("global_load_async_to_lds_b128 %0, %1, off"
                 :: "v"(lds_off), "v"(gaddr) : "memory");
}
__device__ __forceinline__ void wait_async0() {
    asm volatile("s_wait_asynccnt 0x0" ::: "memory");
}

// ---------------------------------------------------------------------------
// Fragment loaders from LDS tiles (layouts per CDNA5 ISA 7.12.2, wave32)
// A 16x32 bf16: lane<16 -> row r0+l, K {0..7, 16..23}; lane>=16 -> K {8..15, 24..31}
// B 32x16 bf16 (from B^T stored [N,K]): lane<16 -> col n0+l, K {0..7, 16..23};
//   lane>=16 -> K {8..15, 24..31}
// ---------------------------------------------------------------------------
__device__ __forceinline__ v16bf frag_A_lds(const __bf16* tile, int row0, int ln, int lh) {
    const __bf16* p = tile + (row0 + ln) * LDS_ROW + lh * 8;
    v8bf c0 = *reinterpret_cast<const v8bf*>(p);
    v8bf c1 = *reinterpret_cast<const v8bf*>(p + 16);
    return __builtin_shufflevector(c0, c1, 0,1,2,3,4,5,6,7,8,9,10,11,12,13,14,15);
}
__device__ __forceinline__ v16bf frag_B_lds(const __bf16* tile, int col0, int ln, int lh) {
    const __bf16* p = tile + (col0 + ln) * LDS_ROW + lh * 8;
    v8bf c0 = *reinterpret_cast<const v8bf*>(p);
    v8bf c1 = *reinterpret_cast<const v8bf*>(p + 16);
    return __builtin_shufflevector(c0, c1, 0,1,2,3,4,5,6,7,8,9,10,11,12,13,14,15);
}

// ---------------------------------------------------------------------------
// Kernel 1: per-row L2-normalize x, emit bf16 hi/lo split
// ---------------------------------------------------------------------------
__global__ __launch_bounds__(256)
void nm_normalize_split(const float* __restrict__ x,
                        __bf16* __restrict__ xh, __bf16* __restrict__ xl) {
    const int row = blockIdx.x;
    const float* xr = x + (size_t)row * IN_DIM;
    __shared__ float wsum[8];

    float s = 0.f;
    for (int c = threadIdx.x; c < IN_DIM; c += 256) {
        float v = xr[c];
        s += v * v;
    }
    for (int off = 16; off > 0; off >>= 1)
        s += __shfl_down(s, off, 32);
    if ((threadIdx.x & 31) == 0) wsum[threadIdx.x >> 5] = s;
    __syncthreads();
    if (threadIdx.x == 0) {
        float t = 0.f;
        for (int w = 0; w < 8; ++w) t += wsum[w];
        wsum[0] = 1.0f / fmaxf(sqrtf(t), 1e-12f);
    }
    __syncthreads();
    const float scale = wsum[0];

    __bf16* hr = xh + (size_t)row * IN_DIM;
    __bf16* lr = xl + (size_t)row * IN_DIM;
    for (int c = threadIdx.x; c < IN_DIM; c += 256) {
        float v  = xr[c] * scale;
        __bf16 h = (__bf16)v;
        hr[c] = h;
        lr[c] = (__bf16)(v - (float)h);
    }
}

// ---------------------------------------------------------------------------
// Kernel 2: elementwise bf16 hi/lo split of a fp32 weight matrix
// ---------------------------------------------------------------------------
__global__ __launch_bounds__(256)
void nm_split_bf16(const float* __restrict__ src,
                   __bf16* __restrict__ hi, __bf16* __restrict__ lo, int n) {
    for (int i = blockIdx.x * 256 + threadIdx.x; i < n; i += gridDim.x * 256) {
        float v  = src[i];
        __bf16 h = (__bf16)v;
        hi[i] = h;
        lo[i] = (__bf16)(v - (float)h);
    }
}

// ---------------------------------------------------------------------------
// Kernel 3: bf16x3 WMMA GEMM with async-LDS double buffering.
//   C[M,N] = (Ah+Al)[M,K] * (Bh+Bl)[N,K]^T, acc = hh + hl + lh in fp32.
//   256 threads = 8 waves as 2(M) x 4(N); block tile 64x128, wave tile 32x32.
//   fuse==1: out[m,n] = xres[m,n] + gate[n]*acc  (decode epilogue)
// ---------------------------------------------------------------------------
__global__ __launch_bounds__(256)
void nm_gemm_bf16x3(const __bf16* __restrict__ Ah, const __bf16* __restrict__ Al,
                    const __bf16* __restrict__ Bh, const __bf16* __restrict__ Bl,
                    int M, int N, int Kd,
                    float* __restrict__ Cout,
                    const float* __restrict__ xres, const float* __restrict__ gate,
                    float* __restrict__ outp, int fuse) {
    __shared__ __bf16 sAh[2][BM * LDS_ROW];   // 2 x 5120 B
    __shared__ __bf16 sAl[2][BM * LDS_ROW];
    __shared__ __bf16 sBh[2][BN * LDS_ROW];   // 2 x 10240 B
    __shared__ __bf16 sBl[2][BN * LDS_ROW];

    const int t    = threadIdx.x;
    const int lane = t & 31;
    const int ln   = lane & 15;
    const int lh   = lane >> 4;
    const int wave = t >> 5;                // 0..7
    const int wm   = wave >> 2;             // 0..1
    const int wn   = wave & 3;              // 0..3

    const int mblk = blockIdx.y * BM;
    const int nblk = blockIdx.x * BN;
    const int mw   = wm * 32;               // wave M origin within block
    const int nw   = wn * 32;               // wave N origin within block

    // LDS base byte offsets for async DMA (low 32 bits of generic pointer)
    unsigned ldsAh[2], ldsAl[2], ldsBh[2], ldsBl[2];
    for (int p = 0; p < 2; ++p) {
        ldsAh[p] = (unsigned)(uintptr_t)&sAh[p][0];
        ldsAl[p] = (unsigned)(uintptr_t)&sAl[p][0];
        ldsBh[p] = (unsigned)(uintptr_t)&sBh[p][0];
        ldsBl[p] = (unsigned)(uintptr_t)&sBl[p][0];
    }

    // Per-thread DMA assignment: 16-byte chunks (8 bf16). A: 64 rows x 4 chunks
    // = 256 chunks (1/thread). B: 128 rows x 4 chunks = 512 chunks (2/thread).
    const int ar = t >> 2, ac = t & 3;
    const int br0 = t >> 1 >> 1, bc0 = t & 3;            // chunk t
    const int br1 = (t + 256) >> 2, bc1 = (t + 256) & 3; // chunk t+256

    const int nk = Kd / BK;

#define ISSUE_TILE(p, kk)                                                          \
    do {                                                                           \
        async_b128(ldsAh[p] + (unsigned)(ar * LDS_ROW + ac * 8) * 2,               \
            (unsigned long long)(uintptr_t)(Ah + (size_t)(mblk + ar) * Kd + (kk) + ac * 8)); \
        async_b128(ldsAl[p] + (unsigned)(ar * LDS_ROW + ac * 8) * 2,               \
            (unsigned long long)(uintptr_t)(Al + (size_t)(mblk + ar) * Kd + (kk) + ac * 8)); \
        async_b128(ldsBh[p] + (unsigned)(br0 * LDS_ROW + bc0 * 8) * 2,             \
            (unsigned long long)(uintptr_t)(Bh + (size_t)(nblk + br0) * Kd + (kk) + bc0 * 8)); \
        async_b128(ldsBl[p] + (unsigned)(br0 * LDS_ROW + bc0 * 8) * 2,             \
            (unsigned long long)(uintptr_t)(Bl + (size_t)(nblk + br0) * Kd + (kk) + bc0 * 8)); \
        async_b128(ldsBh[p] + (unsigned)(br1 * LDS_ROW + bc1 * 8) * 2,             \
            (unsigned long long)(uintptr_t)(Bh + (size_t)(nblk + br1) * Kd + (kk) + bc1 * 8)); \
        async_b128(ldsBl[p] + (unsigned)(br1 * LDS_ROW + bc1 * 8) * 2,             \
            (unsigned long long)(uintptr_t)(Bl + (size_t)(nblk + br1) * Kd + (kk) + bc1 * 8)); \
    } while (0)

    v8f acc[2][2];
#pragma unroll
    for (int i = 0; i < 2; ++i)
#pragma unroll
        for (int j = 0; j < 2; ++j)
            acc[i][j] = (v8f)(0.0f);

    // Prologue: stage slab 0
    ISSUE_TILE(0, 0);
    wait_async0();
    __syncthreads();

    for (int kt = 0; kt < nk; ++kt) {
        const int p = kt & 1;
        if (kt + 1 < nk) ISSUE_TILE(p ^ 1, (kt + 1) * BK);  // overlap DMA with WMMA

        v16bf fah[2], fal[2], fbh[2], fbl[2];
#pragma unroll
        for (int i = 0; i < 2; ++i) {
            fah[i] = frag_A_lds(sAh[p], mw + i * 16, ln, lh);
            fal[i] = frag_A_lds(sAl[p], mw + i * 16, ln, lh);
        }
#pragma unroll
        for (int j = 0; j < 2; ++j) {
            fbh[j] = frag_B_lds(sBh[p], nw + j * 16, ln, lh);
            fbl[j] = frag_B_lds(sBl[p], nw + j * 16, ln, lh);
        }
#pragma unroll
        for (int i = 0; i < 2; ++i) {
#pragma unroll
            for (int j = 0; j < 2; ++j) {
                acc[i][j] = __builtin_amdgcn_wmma_f32_16x16x32_bf16(
                    false, fah[i], false, fbh[j], (short)0, acc[i][j], false, false);
                acc[i][j] = __builtin_amdgcn_wmma_f32_16x16x32_bf16(
                    false, fah[i], false, fbl[j], (short)0, acc[i][j], false, false);
                acc[i][j] = __builtin_amdgcn_wmma_f32_16x16x32_bf16(
                    false, fal[i], false, fbh[j], (short)0, acc[i][j], false, false);
            }
        }

        wait_async0();       // slab kt+1 fully landed in LDS
        __syncthreads();     // all waves done reading slab kt
    }
#undef ISSUE_TILE

    // C/D layout (ISA 7.12.2): VGPR j, lane<16 -> (M=j, N=lane); lane>=16 -> (M=8+j)
#pragma unroll
    for (int i = 0; i < 2; ++i) {
#pragma unroll
        for (int j = 0; j < 2; ++j) {
            const int n = nblk + nw + j * 16 + ln;
#pragma unroll
            for (int vg = 0; vg < 8; ++vg) {
                const int m = mblk + mw + i * 16 + lh * 8 + vg;
                const float v = acc[i][j][vg];
                const size_t idx = (size_t)m * N + n;
                if (fuse)
                    outp[idx] = xres[idx] + gate[n] * v;
                else
                    Cout[idx] = v;
            }
        }
    }
}

// ---------------------------------------------------------------------------
// Kernel 4: per-row top-K threshold (radix-select over order-preserving u32
// keys, MSB-first 8-bit passes, histograms in LDS) + masked bf16 hi/lo split.
// ---------------------------------------------------------------------------
__global__ __launch_bounds__(256)
void nm_topk_mask_split(const float* __restrict__ hidden,
                        __bf16* __restrict__ acth, __bf16* __restrict__ actl) {
    const int row = blockIdx.x;
    const float* hr = hidden + (size_t)row * HID_DIM;

    __shared__ unsigned keys[HID_DIM];
    __shared__ unsigned hist[256];
    __shared__ unsigned sPrefix, sRemain;

    for (int c = threadIdx.x; c < HID_DIM; c += 256) {
        unsigned b = __float_as_uint(hr[c]);
        keys[c] = (b & 0x80000000u) ? ~b : (b | 0x80000000u);
    }
    if (threadIdx.x == 0) { sPrefix = 0u; sRemain = K_SEL; }
    __syncthreads();

    for (int p = 24; p >= 0; p -= 8) {
        const unsigned himask = (p == 24) ? 0u : (0xFFFFFFFFu << (p + 8));
        const unsigned prefix = sPrefix;

        hist[threadIdx.x] = 0u;
        __syncthreads();

        for (int c = threadIdx.x; c < HID_DIM; c += 256) {
            const unsigned k = keys[c];
            if ((k & himask) == (prefix & himask))
                atomicAdd(&hist[(k >> p) & 0xFFu], 1u);
        }
        __syncthreads();

        if (threadIdx.x == 0) {
            unsigned remain = sRemain, cum = 0u;
            int sel = 0;
            for (int bin = 255; bin >= 0; --bin) {
                const unsigned c = hist[bin];
                if (cum + c >= remain) { sel = bin; break; }
                cum += c;
            }
            sPrefix = prefix | ((unsigned)sel << p);
            sRemain = remain - cum;
        }
        __syncthreads();
    }

    const unsigned T = sPrefix;
    __bf16* ah = acth + (size_t)row * HID_DIM;
    __bf16* al = actl + (size_t)row * HID_DIM;
    for (int c = threadIdx.x; c < HID_DIM; c += 256) {
        const unsigned u = keys[c];
        float v = 0.0f;
        if (u >= T) {
            const unsigned b = (u & 0x80000000u) ? (u ^ 0x80000000u) : ~u;
            v = __uint_as_float(b);
        }
        __bf16 h = (__bf16)v;
        ah[c] = h;
        al[c] = (__bf16)(v - (float)h);
    }
}

// ---------------------------------------------------------------------------
// Launcher
// ---------------------------------------------------------------------------
extern "C" void kernel_launch(void* const* d_in, const int* in_sizes, int n_in,
                              void* d_out, int out_size, void* d_ws, size_t ws_size,
                              hipStream_t stream) {
    const float* x     = (const float*)d_in[0];   // [4096,1024]
    const float* W_enc = (const float*)d_in[1];   // [8192,1024]
    const float* W_dec = (const float*)d_in[2];   // [1024,8192]
    const float* gate  = (const float*)d_in[3];   // [1024]
    float* out = (float*)d_out;                   // [4096,1024]

    char* ws = (char*)d_ws;
    size_t off = 0;
    float*  hidden = (float*)(ws + off);  off += (size_t)B_DIM * HID_DIM * 4;   // 128 MB
    __bf16* xh     = (__bf16*)(ws + off); off += (size_t)B_DIM * IN_DIM * 2;    //   8 MB
    __bf16* xl     = (__bf16*)(ws + off); off += (size_t)B_DIM * IN_DIM * 2;
    __bf16* weh    = (__bf16*)(ws + off); off += (size_t)HID_DIM * IN_DIM * 2;  //  16 MB
    __bf16* wel    = (__bf16*)(ws + off); off += (size_t)HID_DIM * IN_DIM * 2;
    __bf16* wdh    = (__bf16*)(ws + off); off += (size_t)IN_DIM * HID_DIM * 2;
    __bf16* wdl    = (__bf16*)(ws + off); off += (size_t)IN_DIM * HID_DIM * 2;
    __bf16* acth   = (__bf16*)(ws + off); off += (size_t)B_DIM * HID_DIM * 2;   //  64 MB
    __bf16* actl   = (__bf16*)(ws + off); off += (size_t)B_DIM * HID_DIM * 2;

    nm_normalize_split<<<B_DIM, 256, 0, stream>>>(x, xh, xl);
    nm_split_bf16<<<8192, 256, 0, stream>>>(W_enc, weh, wel, HID_DIM * IN_DIM);
    nm_split_bf16<<<8192, 256, 0, stream>>>(W_dec, wdh, wdl, IN_DIM * HID_DIM);

    {   // encode GEMM: hidden[B,HID] = x_norm @ W_enc^T
        dim3 grid(HID_DIM / BN, B_DIM / BM);
        nm_gemm_bf16x3<<<grid, 256, 0, stream>>>(xh, xl, weh, wel,
                                                 B_DIM, HID_DIM, IN_DIM,
                                                 hidden, nullptr, nullptr, nullptr, 0);
    }

    nm_topk_mask_split<<<B_DIM, 256, 0, stream>>>(hidden, acth, actl);

    {   // decode GEMM fused with residual: out = x + gate * (act @ W_dec^T)
        dim3 grid(IN_DIM / BN, B_DIM / BM);
        nm_gemm_bf16x3<<<grid, 256, 0, stream>>>(acth, actl, wdh, wdl,
                                                 B_DIM, IN_DIM, HID_DIM,
                                                 nullptr, x, gate, out, 1);
    }
}
